// PolicyNetwork_2396591751191
// MI455X (gfx1250) — compile-verified, matrix operand order checked by reading, not run on hardware
//
#include <hip/hip_runtime.h>
#include <hip/hip_bf16.h>

// ---------------------------------------------------------------------------
// MI455X (gfx1250) implementation. wave32, WMMA f32_16x16x32_f16 everywhere.
// ---------------------------------------------------------------------------

typedef __attribute__((ext_vector_type(16))) _Float16 v16h;
typedef __attribute__((ext_vector_type(8)))  float    v8f;

#define GS(i, n) for (size_t i = (size_t)blockIdx.x * blockDim.x + threadIdx.x; \
                      i < (size_t)(n); i += (size_t)gridDim.x * blockDim.x)

static constexpr int NN = 4096;   // nodes
static constexpr int HH = 128;    // hidden
static constexpr int TT = 32;     // timesteps
static constexpr int EE = 65536;  // edges

__device__ inline v8f zero8() {
    v8f z;
#pragma unroll
    for (int i = 0; i < 8; i++) z[i] = 0.f;
    return z;
}

__device__ inline v8f wmma_f16(v16h a, v16h b, v8f c) {
    // 8 args: (neg_a, A, neg_b, B, c_mod, C, reuse_a, reuse_b)
    return __builtin_amdgcn_wmma_f32_16x16x32_f16(false, a, false, b, (short)0, c,
                                                  false, false);
}

// Load one 16x32(A) / 32x16(B) f16 fragment for this lane with two 16B loads.
// p points at this lane's 32-element K-chunk; must be 16B aligned.
// Lane layout (ISA 7.12.2, 16-bit A 16x32):
//   lanes 0-15 : K = 0..7 then 16..23 ; lanes 16-31: K = 8..15 then 24..31
__device__ inline v16h frag32(const _Float16* p, int lane) {
    int kb = (lane & 16) ? 8 : 0;
    v16h f;
    *((float4*)&f)     = *(const float4*)(p + kb);       // K-chunk elems 0..7
    *((float4*)&f + 1) = *(const float4*)(p + kb + 16);  // K-chunk elems 16..23
    return f;
}

__device__ inline float sigmoidf_(float x) { return 1.f / (1.f + expf(-x)); }

__device__ inline float atomicMaxF(float* a, float v) {
    int old = __float_as_int(*a);
    while (true) {
        if (__int_as_float(old) >= v) break;
        int prev = atomicCAS((int*)a, old, __float_as_int(v));
        if (prev == old) break;
        old = prev;
    }
    return __int_as_float(old);
}

// ---------------------------------------------------------------------------
// Conversion helpers
// ---------------------------------------------------------------------------
__global__ void cvt16_kernel(const float* in, _Float16* out, size_t n) {
    GS(i, n) out[i] = (_Float16)in[i];
}

// W [K][Nc] (row-major, x@W) -> Wt [Nc][K] f16 (so B-fragment loads are contiguous)
__global__ void cvtT16_kernel(const float* W, _Float16* Wt, int K, int Nc) {
    GS(i, (size_t)K * Nc) {
        int n = (int)(i / K);
        int k = (int)(i - (size_t)n * K);
        Wt[i] = (_Float16)W[(size_t)k * Nc + n];
    }
}

// Wih [384][16] -> [384][32] f16 zero-padded in K (K=16 padded to one WMMA step)
__global__ void cvt_wih_kernel(const float* Wih, _Float16* o) {
    GS(i, 384 * 32) {
        int n = (int)(i >> 5), k = (int)(i & 31);
        o[i] = (k < 16) ? (_Float16)Wih[n * 16 + k] : (_Float16)0.f;
    }
}

// pos16[i][j] = f16(eig[sub[i]][j])
__global__ void build_pos16_kernel(const float* eig, const int* sub, _Float16* pos16) {
    GS(i, (size_t)NN * NN) {
        int r = (int)(i >> 12);
        int c = (int)(i & (NN - 1));
        pos16[i] = (_Float16)eig[(size_t)sub[r] * NN + c];
    }
}

__global__ void gather_x16_kernel(const _Float16* h16, const int* sub, _Float16* out) {
    GS(i, (size_t)NN * HH) {
        int r = (int)(i >> 7), k = (int)(i & 127);
        out[i] = h16[(size_t)sub[r] * HH + k];
    }
}

// xcat = [x16 (128) | prev16 (128, optional) | pos16 (4096)]
__global__ void build_xcat_kernel(const _Float16* x16, const _Float16* p16,
                                  const _Float16* pos16, _Float16* xcat,
                                  int K, int hasprev) {
    GS(i, (size_t)NN * K) {
        int row = (int)(i / K);
        int k   = (int)(i - (size_t)row * K);
        _Float16 v;
        if (k < 128)                 v = x16[(size_t)row * 128 + k];
        else if (hasprev && k < 256) v = p16[(size_t)row * 128 + (k - 128)];
        else {
            int off = hasprev ? 256 : 128;
            v = pos16[(size_t)row * NN + (k - off)];
        }
        xcat[i] = v;
    }
}

// ---------------------------------------------------------------------------
// GRU: one wave per 16 nodes, h kept in LDS, WMMA for all gate GEMMs.
// Weights stream from L2-resident f16 buffers; the jt loop is rolled (keeps
// live fragments bounded -> no scratch spills) while the 4-step K-chunk loop
// is unrolled (constant indices into ah[] -> no v_movrels).
// ---------------------------------------------------------------------------
__global__ __launch_bounds__(32) void gru_kernel(
    const float* __restrict__ x, const _Float16* __restrict__ wih16,
    const _Float16* __restrict__ whh16, const float* __restrict__ bih,
    const float* __restrict__ bhh, float* __restrict__ h32,
    _Float16* __restrict__ h16) {
    __shared__ __align__(16) _Float16 sH[16 * 128];
    int lane = threadIdx.x & 31;
    int node = blockIdx.x * 16 + (lane & 15);

    for (int i = lane; i < 16 * 128 / 2; i += 32) ((uint*)sH)[i] = 0u;
    __syncthreads();

    int kb   = (lane & 16) ? 8 : 0;
    int lrow = (lane & 16) ? 8 : 0;

#pragma clang loop unroll(disable)
    for (int t = 0; t < TT; t++) {
        // A fragment from x_t (K=16 real, 16..31 zero-pad)
        const float* xp = x + (((size_t)t * NN) + node) * 16 + kb;
        v16h ax;
#pragma unroll
        for (int i = 0; i < 8; i++) {
            ax[i]     = (kb + i < 16) ? (_Float16)xp[i] : (_Float16)0.f;
            ax[8 + i] = (_Float16)0.f;  // K 16..31 padding
        }
        // A fragments from h (K = 128 -> 4 chunks); captured BEFORE any write
        const _Float16* hp = sH + (size_t)(lane & 15) * 128;
        v16h ah0 = frag32(hp + 0 * 32, lane);
        v16h ah1 = frag32(hp + 1 * 32, lane);
        v16h ah2 = frag32(hp + 2 * 32, lane);
        v16h ah3 = frag32(hp + 3 * 32, lane);

#pragma clang loop unroll(disable)
        for (int jt = 0; jt < 8; jt++) {
            int hcol = jt * 16 + (lane & 15);
            v8f ar = zero8(), az = zero8(), ain = zero8(), ahn = zero8();
            // x contributions (1 WMMA each)
            ar  = wmma_f16(ax, frag32(wih16 + (size_t)(0   + hcol) * 32, lane), ar);
            az  = wmma_f16(ax, frag32(wih16 + (size_t)(128 + hcol) * 32, lane), az);
            ain = wmma_f16(ax, frag32(wih16 + (size_t)(256 + hcol) * 32, lane), ain);
            // h contributions: 4 K-chunks, constant-indexed A fragments
            const _Float16* wr = whh16 + (size_t)(0   + hcol) * 128;
            const _Float16* wz = whh16 + (size_t)(128 + hcol) * 128;
            const _Float16* wn = whh16 + (size_t)(256 + hcol) * 128;
            ar  = wmma_f16(ah0, frag32(wr + 0 * 32, lane), ar);
            az  = wmma_f16(ah0, frag32(wz + 0 * 32, lane), az);
            ahn = wmma_f16(ah0, frag32(wn + 0 * 32, lane), ahn);
            ar  = wmma_f16(ah1, frag32(wr + 1 * 32, lane), ar);
            az  = wmma_f16(ah1, frag32(wz + 1 * 32, lane), az);
            ahn = wmma_f16(ah1, frag32(wn + 1 * 32, lane), ahn);
            ar  = wmma_f16(ah2, frag32(wr + 2 * 32, lane), ar);
            az  = wmma_f16(ah2, frag32(wz + 2 * 32, lane), az);
            ahn = wmma_f16(ah2, frag32(wn + 2 * 32, lane), ahn);
            ar  = wmma_f16(ah3, frag32(wr + 3 * 32, lane), ar);
            az  = wmma_f16(ah3, frag32(wz + 3 * 32, lane), az);
            ahn = wmma_f16(ah3, frag32(wn + 3 * 32, lane), ahn);

            float br = bih[hcol], bz = bih[128 + hcol], bn = bih[256 + hcol];
            float cr = bhh[hcol], cz = bhh[128 + hcol], cn = bhh[256 + hcol];
            // gates + write h tile straight back to LDS (A-frags already in regs)
#pragma unroll
            for (int i = 0; i < 8; i++) {
                float r  = sigmoidf_(ar[i] + br + cr);
                float zg = sigmoidf_(az[i] + bz + cz);
                float nv = tanhf(ain[i] + bn + r * (ahn[i] + cn));
                float ho = (float)sH[(size_t)(lrow + i) * 128 + hcol];
                sH[(size_t)(lrow + i) * 128 + hcol] =
                    (_Float16)((1.f - zg) * nv + zg * ho);
            }
        }
        __syncthreads();
    }
    // write out final h
    for (int i = lane; i < 16 * 128; i += 32) {
        int lm = i >> 7, c = i & 127;
        int gn = blockIdx.x * 16 + lm;
        float v = (float)sH[i];
        h32[(size_t)gn * 128 + c] = v;
        h16[(size_t)gn * 128 + c] = (_Float16)v;
    }
}

// ---------------------------------------------------------------------------
// Generic f16 WMMA GEMM: C[M][Nc] = A[M][K] * Wt[Nc][K]^T + bias
// 128x128 tile / block of 256 threads = 8 waves; each wave a 16x128 strip.
// ---------------------------------------------------------------------------
__global__ __launch_bounds__(256) void gemm_f16_kernel(
    const _Float16* __restrict__ A, const _Float16* __restrict__ B,
    const float* __restrict__ bias, float* __restrict__ C,
    int M, int Nc, int K) {
    __shared__ __align__(16) _Float16 sA[128 * 32];
    __shared__ __align__(16) _Float16 sB[128 * 32];
    int tid = threadIdx.x, lane = tid & 31, wave = tid >> 5;
    int m0 = blockIdx.x * 128, n0 = blockIdx.y * 128;

    v8f acc[8];
#pragma unroll
    for (int i = 0; i < 8; i++) acc[i] = zero8();

    // each thread stages one 32B chunk of A and of B per K-step
    int srow = tid >> 1;              // 0..127
    int scu  = (tid & 1) * 8;         // uint offset within row (0 or 8)

    for (int kbase = 0; kbase < K; kbase += 32) {
        {
            const uint4* gA = (const uint4*)(A + (size_t)(m0 + srow) * K + kbase + scu * 2);
            const uint4* gB = (const uint4*)(B + (size_t)(n0 + srow) * K + kbase + scu * 2);
            uint4 a0 = gA[0], a1 = gA[1];
            uint4 b0 = gB[0], b1 = gB[1];
            ((uint4*)sA)[tid * 2 + 0] = a0;
            ((uint4*)sA)[tid * 2 + 1] = a1;
            ((uint4*)sB)[tid * 2 + 0] = b0;
            ((uint4*)sB)[tid * 2 + 1] = b1;
        }
        __syncthreads();
        v16h af = frag32(sA + (size_t)(wave * 16 + (lane & 15)) * 32, lane);
#pragma unroll
        for (int nt = 0; nt < 8; nt++) {
            v16h bf = frag32(sB + (size_t)(nt * 16 + (lane & 15)) * 32, lane);
            acc[nt] = wmma_f16(af, bf, acc[nt]);
        }
        __syncthreads();
    }
    int mrow = m0 + wave * 16 + ((lane & 16) ? 8 : 0);
#pragma unroll
    for (int nt = 0; nt < 8; nt++) {
        int n = n0 + nt * 16 + (lane & 15);
        float b = bias[n];
#pragma unroll
        for (int i = 0; i < 8; i++)
            C[(size_t)(mrow + i) * Nc + n] = acc[nt][i] + b;
    }
}

// ---------------------------------------------------------------------------
// Attention (per-destination softmax over edges)
// ---------------------------------------------------------------------------
__global__ void init_mdenom_kernel(float* m, float* d, int n) {
    GS(i, n) { m[i] = -1e30f; d[i] = 0.f; }
}

__global__ void att_score_kernel(const float* __restrict__ q,
                                 const float* __restrict__ k,
                                 const int* __restrict__ ei,
                                 float* __restrict__ score, float* __restrict__ mbuf,
                                 int E, int heads, int HD) {
    GS(idx, (size_t)E * heads) {
        int e = (int)(idx / heads), h = (int)(idx % heads);
        int src = ei[e], dst = ei[E + e];
        const float4* qp = (const float4*)(q + (size_t)dst * HD + h * 128);
        const float4* kp = (const float4*)(k + (size_t)src * HD + h * 128);
        float s = 0.f;
#pragma unroll 8
        for (int d = 0; d < 32; d++) {
            float4 a = qp[d], b = kp[d];
            s += a.x * b.x + a.y * b.y + a.z * b.z + a.w * b.w;
        }
        s *= 0.08838834764831845f;  // 1/sqrt(128)
        score[idx] = s;
        atomicMaxF(&mbuf[dst * heads + h], s);
    }
}

__global__ void att_exp_kernel(const float* __restrict__ score,
                               const float* __restrict__ mbuf,
                               float* __restrict__ denom, float* __restrict__ exb,
                               const int* __restrict__ ei, int E, int heads) {
    GS(idx, (size_t)E * heads) {
        int e = (int)(idx / heads), h = (int)(idx % heads);
        int dst = ei[E + e];
        float ex = expf(score[idx] - mbuf[dst * heads + h]);
        exb[idx] = ex;
        atomicAdd(&denom[dst * heads + h], ex);
    }
}

__global__ void att_agg_kernel(const float* __restrict__ exb,
                               const float* __restrict__ denom,
                               const float* __restrict__ v,
                               const int* __restrict__ ei, float* __restrict__ out,
                               int E, int heads, int HD) {
    GS(idx, (size_t)E * heads * 128) {
        int d = (int)(idx & 127);
        size_t eh = idx >> 7;
        int e = (int)(eh / heads), h = (int)(eh % heads);
        int src = ei[e], dst = ei[E + e];
        float alpha = exb[eh] / denom[dst * heads + h];
        atomicAdd(&out[(size_t)dst * HD + h * 128 + d],
                  alpha * v[(size_t)src * HD + h * 128 + d]);
    }
}

// ---------------------------------------------------------------------------
// BatchNorm (train mode, biased var) + LeakyReLU -> f16
// ---------------------------------------------------------------------------
__global__ __launch_bounds__(256) void bn_stats_kernel(const float* __restrict__ y,
                                                       float* mean, float* inv) {
    int c = blockIdx.x;  // 0..127
    __shared__ float ss[256], sq[256];
    int tid = threadIdx.x;
    float s = 0.f, q2 = 0.f;
    for (int r = tid; r < NN; r += 256) {
        float v = y[(size_t)r * 128 + c];
        s += v; q2 += v * v;
    }
    ss[tid] = s; sq[tid] = q2;
    __syncthreads();
    for (int o = 128; o > 0; o >>= 1) {
        if (tid < o) { ss[tid] += ss[tid + o]; sq[tid] += sq[tid + o]; }
        __syncthreads();
    }
    if (tid == 0) {
        float mu  = ss[0] / (float)NN;
        float var = sq[0] / (float)NN - mu * mu;
        mean[c] = mu;
        inv[c]  = rsqrtf(var + 1e-5f);
    }
}

__global__ void bn_apply_kernel(const float* __restrict__ y, const float* mean,
                                const float* inv, const float* gamma,
                                const float* beta, _Float16* __restrict__ out16) {
    GS(i, (size_t)NN * 128) {
        int c = (int)(i & 127);
        float v = (y[i] - mean[c]) * inv[c] * gamma[c] + beta[c];
        v = (v > 0.f) ? v : 0.01f * v;  // LeakyReLU
        out16[i] = (_Float16)v;
    }
}

// ---------------------------------------------------------------------------
// Projection head on the agent's row + action mask
// ---------------------------------------------------------------------------
__global__ __launch_bounds__(128) void head_kernel(
    const float* __restrict__ xfin, const float* __restrict__ W1,
    const float* __restrict__ b1, const float* __restrict__ W2,
    const float* __restrict__ b2, const float* __restrict__ mask,
    const int* __restrict__ agentp, const int* __restrict__ sub,
    float* __restrict__ out) {
    __shared__ float sz[128], sz1[128];
    __shared__ int sloc;
    int tid = threadIdx.x;
    if (tid == 0) {
        int a = agentp[0], loc = 0;
        for (int i = 0; i < NN; i++) { if (sub[i] == a) { loc = i; break; } }
        sloc = loc;
    }
    __syncthreads();
    sz[tid] = xfin[(size_t)sloc * 128 + tid];
    __syncthreads();
    float acc = b1[tid];
    for (int k2 = 0; k2 < 128; k2++) acc += sz[k2] * W1[k2 * 128 + tid];
    sz1[tid] = (acc > 0.f) ? acc : 0.01f * acc;
    __syncthreads();
    if (tid < 8) {
        int a = agentp[0];
        float lg = b2[tid];
        for (int j = 0; j < 128; j++) lg += sz1[j] * W2[j * 8 + tid];
        out[tid] = lg + (1.f - mask[(size_t)a * 8 + tid]) * (-1e9f);
    }
}

// ---------------------------------------------------------------------------
// Host orchestration
// ---------------------------------------------------------------------------
extern "C" void kernel_launch(void* const* d_in, const int* in_sizes, int n_in,
                              void* d_out, int out_size, void* d_ws, size_t ws_size,
                              hipStream_t stream) {
    (void)in_sizes; (void)n_in; (void)out_size; (void)ws_size;
    // d_in index map (setup_inputs dict order, params flattened depth-first)
    enum {
        IN_X = 0, IN_EIG = 1, IN_MASK = 2,
        IN_WIH = 3, IN_WHH = 4, IN_BIH = 5, IN_BHH = 6,
        L0 = 7,   // Wq bq Wk bk Wv bv Ws bs Wp bp gamma beta
        L1 = 19,
        L2 = 31,  // conv only: Wq bq Wk bk Wv bv Ws bs
        PH_W1 = 39, PH_B1 = 40, PH_W2 = 41, PH_B2 = 42,
        IN_EI = 43, IN_AGENT = 44, IN_SUB = 45,
    };

    char* ws = (char*)d_ws;
    size_t off = 0;
    auto alloc = [&](size_t bytes) {
        size_t o = off;
        off = (off + bytes + 255) & ~(size_t)255;
        return (void*)(ws + o);
    };
    float*     h32    = (float*)alloc((size_t)NN * HH * 4);
    _Float16*  h16    = (_Float16*)alloc((size_t)NN * HH * 2);
    _Float16*  pos16  = (_Float16*)alloc((size_t)NN * NN * 2);
    _Float16*  x16a   = (_Float16*)alloc((size_t)NN * HH * 2);
    _Float16*  x16b   = (_Float16*)alloc((size_t)NN * HH * 2);
    _Float16*  x16c   = (_Float16*)alloc((size_t)NN * HH * 2);
    _Float16*  xcat   = (_Float16*)alloc((size_t)NN * 4352 * 2);
    _Float16*  wt16   = (_Float16*)alloc((size_t)512 * 4352 * 2);
    _Float16*  wihg   = (_Float16*)alloc((size_t)384 * 32 * 2);
    _Float16*  whhg   = (_Float16*)alloc((size_t)384 * 128 * 2);
    float*     qb     = (float*)alloc((size_t)NN * 512 * 4);
    float*     kbuf   = (float*)alloc((size_t)NN * 512 * 4);
    float*     vb     = (float*)alloc((size_t)NN * 512 * 4);
    float*     aggb   = (float*)alloc((size_t)NN * 512 * 4);
    _Float16*  conv16 = (_Float16*)alloc((size_t)NN * 512 * 2);
    float*     yb     = (float*)alloc((size_t)NN * 128 * 4);
    float*     score  = (float*)alloc((size_t)EE * 4 * 4);
    float*     exb    = (float*)alloc((size_t)EE * 4 * 4);
    float*     mb     = (float*)alloc((size_t)NN * 4 * 4);
    float*     db     = (float*)alloc((size_t)NN * 4 * 4);
    float*     bnm    = (float*)alloc(128 * 4);
    float*     bni    = (float*)alloc(128 * 4);

    const float* x    = (const float*)d_in[IN_X];
    const float* eig  = (const float*)d_in[IN_EIG];
    const float* mask = (const float*)d_in[IN_MASK];
    const int*   ei   = (const int*)d_in[IN_EI];
    const int*   ag   = (const int*)d_in[IN_AGENT];
    const int*   sub  = (const int*)d_in[IN_SUB];

    // --- GRU ---
    cvt_wih_kernel<<<48, 256, 0, stream>>>((const float*)d_in[IN_WIH], wihg);
    cvt16_kernel<<<192, 256, 0, stream>>>((const float*)d_in[IN_WHH], whhg, 384 * 128);
    gru_kernel<<<NN / 16, 32, 0, stream>>>(x, wihg, whhg,
                                           (const float*)d_in[IN_BIH],
                                           (const float*)d_in[IN_BHH], h32, h16);

    // --- positional features + layer-0 input ---
    build_pos16_kernel<<<4096, 256, 0, stream>>>(eig, sub, pos16);
    gather_x16_kernel<<<2048, 256, 0, stream>>>(h16, sub, x16a);

    const int  convBase[3] = {L0, L1, L2};
    const int  Kc[3]       = {4224, 4352, 4352};
    const int  HD[3]       = {512, 512, 128};
    const int  heads[3]    = {4, 4, 1};
    const _Float16* curx   = x16a;
    const _Float16* prevx  = nullptr;
    _Float16* nextbuf[2]   = {x16b, x16c};

    for (int l = 0; l < 3; l++) {
        int cb = convBase[l], K = Kc[l], hd = HD[l], nh = heads[l];
        build_xcat_kernel<<<8192, 256, 0, stream>>>(curx, prevx, pos16, xcat, K,
                                                    prevx != nullptr);
        // Q, K, V, S projections (S written directly into agg buffer = skip path)
        float* outs[4] = {qb, kbuf, vb, aggb};
        for (int wsel = 0; wsel < 4; wsel++) {
            const float* W = (const float*)d_in[cb + wsel * 2];
            const float* b = (const float*)d_in[cb + wsel * 2 + 1];
            cvtT16_kernel<<<4096, 256, 0, stream>>>(W, wt16, K, hd);
            gemm_f16_kernel<<<dim3(NN / 128, hd / 128), 256, 0, stream>>>(
                xcat, wt16, b, outs[wsel], NN, hd, K);
        }
        // attention softmax by destination
        init_mdenom_kernel<<<(NN * nh + 255) / 256, 256, 0, stream>>>(mb, db, NN * nh);
        int nEH = EE * nh;
        att_score_kernel<<<(nEH + 255) / 256, 256, 0, stream>>>(qb, kbuf, ei, score,
                                                                mb, EE, nh, hd);
        att_exp_kernel<<<(nEH + 255) / 256, 256, 0, stream>>>(score, mb, db, exb, ei,
                                                              EE, nh);
        att_agg_kernel<<<8192, 256, 0, stream>>>(exb, db, vb, ei, aggb, EE, nh, hd);

        if (l < 2) {
            // proj -> BN -> LeakyReLU -> next layer f16 input
            cvt16_kernel<<<4096, 256, 0, stream>>>(aggb, conv16, (size_t)NN * 512);
            cvtT16_kernel<<<512, 256, 0, stream>>>((const float*)d_in[cb + 8], wt16,
                                                   512, 128);
            gemm_f16_kernel<<<dim3(NN / 128, 1), 256, 0, stream>>>(
                conv16, wt16, (const float*)d_in[cb + 9], yb, NN, 128, 512);
            bn_stats_kernel<<<128, 256, 0, stream>>>(yb, bnm, bni);
            bn_apply_kernel<<<2048, 256, 0, stream>>>(
                yb, bnm, bni, (const float*)d_in[cb + 10],
                (const float*)d_in[cb + 11], nextbuf[l]);
            prevx = curx;
            curx  = nextbuf[l];
        }
    }

    // --- head: x2[local] -> lrelu(W1) -> W2 + action mask ---
    head_kernel<<<1, 128, 0, stream>>>(aggb, (const float*)d_in[PH_W1],
                                       (const float*)d_in[PH_B1],
                                       (const float*)d_in[PH_W2],
                                       (const float*)d_in[PH_B2], mask, ag, sub,
                                       (float*)d_out);
}
